// MultiHeadAttention_3796751089840
// MI455X (gfx1250) — matrix operand order repaired and checked by reading, not compiled, not run on hardware
//
#include <hip/hip_runtime.h>
#include <hip/hip_bf16.h>

// ---------------------------------------------------------------------------
// Types for CDNA5 WMMA (wave32): 16x16x32 bf16 -> f32
// ---------------------------------------------------------------------------
typedef __attribute__((ext_vector_type(16))) __bf16 v16bf;
typedef __attribute__((ext_vector_type(8)))  __bf16 v8bf;
typedef __attribute__((ext_vector_type(8)))  float  v8f;
typedef __attribute__((ext_vector_type(4)))  float  v4f;

static constexpr int Dm    = 1024;
static constexpr int Hh    = 16;
static constexpr int HD    = 64;
static constexpr int Tt    = 2048;
static constexpr int Bb    = 4;
static constexpr int Mrows = Bb * Tt;   // 8192

// Native conversions: gfx1250 has bf16 VALU support; let the backend emit
// packed v_cvt ops instead of manual bfe/add3 rounding sequences.
__device__ __forceinline__ __bf16 f2bf(float x) { return (__bf16)x; }
__device__ __forceinline__ unsigned short f2bfbits(float x) {
    __bf16 h = (__bf16)x;
    return __builtin_bit_cast(unsigned short, h);
}

__device__ __forceinline__ v8f wmma_bf16(v16bf a, v16bf b, v8f c) {
    return __builtin_amdgcn_wmma_f32_16x16x32_bf16(
        false, a, false, b, (short)0, c, false, false);
}

// Load 16 bf16 elements as two contiguous 16-byte chunks.
__device__ __forceinline__ v16bf load_bf16_frag(const unsigned short* p0,
                                                const unsigned short* p1) {
    v8bf lo = *reinterpret_cast<const v8bf*>(p0);
    v8bf hi = *reinterpret_cast<const v8bf*>(p1);
    v16bf r;
#pragma unroll
    for (int i = 0; i < 8; ++i) { r[i] = lo[i]; r[i + 8] = hi[i]; }
    return r;
}

// ---------------------------------------------------------------------------
// Elementwise f32 -> bf16 conversion (vectorized: 8 elems / thread)
// ---------------------------------------------------------------------------
__global__ __launch_bounds__(256) void cvt_kernel(const float* __restrict__ in,
                                                  unsigned short* __restrict__ out) {
    const size_t i8 = (size_t)blockIdx.x * blockDim.x + threadIdx.x;
    v4f f0 = *reinterpret_cast<const v4f*>(in + i8 * 8);
    v4f f1 = *reinterpret_cast<const v4f*>(in + i8 * 8 + 4);
    v8bf o;
#pragma unroll
    for (int i = 0; i < 4; ++i) { o[i] = f2bf(f0[i]); o[i + 4] = f2bf(f1[i]); }
    *reinterpret_cast<v8bf*>(out + i8 * 8) = o;
}

// ---------------------------------------------------------------------------
// Weight convert+transpose: Wt[n][k] = bf16(W[k][n])   (K = N = 1024)
// ---------------------------------------------------------------------------
__global__ __launch_bounds__(256) void wconv_kernel(const float* __restrict__ W,
                                                    unsigned short* __restrict__ Wt) {
    int idx = blockIdx.x * blockDim.x + threadIdx.x;   // 1024*1024 threads
    int k = idx & (Dm - 1);
    int n = idx >> 10;
    Wt[(size_t)n * Dm + k] = f2bfbits(W[(size_t)k * Dm + n]);
}

// ---------------------------------------------------------------------------
// Tiled GEMM:  C[M,N] = A[M,K] * Wt[N,K]^T + bias      (A, Wt bf16)
//   block = 256 threads = 8 waves (4x2), block tile 128x128,
//   wave tile 32x64 = 2x4 WMMA tiles, K step 32.
// OMODE: 0 = bf16 row-major (scale folded), 1 = bf16 transposed-per-head
//        (Vt[b][h][d][t]), 2 = f32 row-major (final output)
// ---------------------------------------------------------------------------
template <int OMODE>
__global__ __launch_bounds__(256) void gemm_kernel(
    const unsigned short* __restrict__ A, const unsigned short* __restrict__ Wt,
    const float* __restrict__ bias, void* __restrict__ Cv, float scale) {
    const int lane  = threadIdx.x & 31;
    const int w     = threadIdx.x >> 5;
    const int wm    = w >> 1;            // 0..3
    const int wn    = w & 1;             // 0..1
    const int lhalf = lane & 15;
    const int lhi   = lane >> 4;
    const int m_base = blockIdx.y * 128 + wm * 32;
    const int n_base = blockIdx.x * 128 + wn * 64;

    v8f acc[2][4] = {};

    for (int k0 = 0; k0 < Dm; k0 += 32) {
        v16bf a[2], b[4];
#pragma unroll
        for (int mt = 0; mt < 2; ++mt) {
            const unsigned short* p =
                A + (size_t)(m_base + mt * 16 + lhalf) * Dm + k0 + lhi * 16;
            a[mt] = load_bf16_frag(p, p + 8);
        }
#pragma unroll
        for (int nt = 0; nt < 4; ++nt) {
            const unsigned short* p =
                Wt + (size_t)(n_base + nt * 16 + lhalf) * Dm + k0 + lhi * 16;
            b[nt] = load_bf16_frag(p, p + 8);
        }
#pragma unroll
        for (int mt = 0; mt < 2; ++mt)
#pragma unroll
            for (int nt = 0; nt < 4; ++nt)
                acc[mt][nt] = wmma_bf16(a[mt], b[nt], acc[mt][nt]);
    }

    // -------- epilogue --------
    const int rbase = lhi * 8;           // lanes 16-31 hold rows M=8..15
#pragma unroll
    for (int mt = 0; mt < 2; ++mt) {
#pragma unroll
        for (int nt = 0; nt < 4; ++nt) {
            const int n  = n_base + nt * 16 + lhalf;
            const float bn = bias[n];
            if (OMODE == 0) {            // bf16 row-major, scale folded
                unsigned short* C = (unsigned short*)Cv;
#pragma unroll
                for (int r = 0; r < 8; ++r) {
                    const int m = m_base + mt * 16 + rbase + r;
                    C[(size_t)m * Dm + n] = f2bfbits((acc[mt][nt][r] + bn) * scale);
                }
            } else if (OMODE == 1) {     // bf16 transposed per head: Vt[b][h][d][t]
                unsigned short* C = (unsigned short*)Cv;
                const int h = n >> 6, d = n & 63;
                const int m0 = m_base + mt * 16 + rbase;
                const int bb = m0 >> 11;         // / 2048
                const int t0 = m0 & 2047;
                v8bf pk;
#pragma unroll
                for (int r = 0; r < 8; ++r) pk[r] = f2bf(acc[mt][nt][r] + bn);
                *reinterpret_cast<v8bf*>(
                    C + (size_t)((bb * Hh + h) * HD + d) * Tt + t0) = pk;
            } else {                     // f32 row-major
                float* C = (float*)Cv;
#pragma unroll
                for (int r = 0; r < 8; ++r) {
                    const int m = m_base + mt * 16 + rbase + r;
                    C[(size_t)m * Dm + n] = acc[mt][nt][r] + bn;
                }
            }
        }
    }
}

// ---------------------------------------------------------------------------
// Causal flash attention in the TRANSPOSED layout.
// Each wave owns 16 query rows of one (b,h); iterates over 64-key tiles.
//   S^T = K * Q^T      (8 WMMA / tile)  -> each q-row lives in ONE lane
//   online softmax     (per-lane scalars; 2 cross-half shuffles per tile)
//   O^T += V^T * P     (8 WMMA / tile)  -> P built in-register from S^T
// qw has (1/sqrt(HD))*log2(e) folded in. vt is [b][h][d][t] bf16.
// ---------------------------------------------------------------------------
__global__ __launch_bounds__(256) void attn_kernel(
    const unsigned short* __restrict__ qw, const unsigned short* __restrict__ kw,
    const unsigned short* __restrict__ vt, unsigned short* __restrict__ attn) {
    const int lane  = threadIdx.x & 31;
    const int w     = threadIdx.x >> 5;
    const int lhalf = lane & 15;
    const int lhi   = lane >> 4;
    const int rbase = lhi * 8;           // this lane's key-row / d-row offset

    const int bh = blockIdx.x;           // 0..63
    const int b  = bh >> 4;
    const int h  = bh & 15;
    const int q0 = blockIdx.y * 128 + w * 16;
    const int hoff = h * HD;
    const int qrow = q0 + lhalf;         // the single query row this lane owns

    // Q^T B-fragments (loop-invariant): lane = column q-row, elems = head dim.
    v16bf bQ[2];
    {
        const unsigned short* qp = qw + (size_t)(b * Tt + qrow) * Dm + hoff;
#pragma unroll
        for (int dc = 0; dc < 2; ++dc) {
            const unsigned short* p = qp + dc * 32 + lhi * 8;
            bQ[dc] = load_bf16_frag(p, p + 16);
        }
    }

    v8f accO[4] = {};                    // O^T: lane = q-row, vgprs = head dims
    float mrun = -3.0e38f, lrun = 0.0f;  // per-lane softmax state (one q-row)

    for (int k0 = 0; k0 < q0 + 16; k0 += 64) {
        const unsigned short* kbase_p =
            kw + (size_t)(b * Tt + k0 + lhalf) * Dm + hoff;
        // ---- S^T = K * Q^T : batch 4 A-frag loads per d-chunk, then WMMA ----
        v8f s[4] = {};
#pragma unroll
        for (int dc = 0; dc < 2; ++dc) {
            v16bf aK[4];
#pragma unroll
            for (int nt = 0; nt < 4; ++nt) {
                const unsigned short* p =
                    kbase_p + (size_t)(nt * 16) * Dm + dc * 32 + lhi * 8;
                aK[nt] = load_bf16_frag(p, p + 16);
            }
#pragma unroll
            for (int nt = 0; nt < 4; ++nt)
                s[nt] = wmma_bf16(aK[nt], bQ[dc], s[nt]);
        }
        // ---- causal mask: element (key, q) masked when key > q ----
        if (k0 + 63 > q0) {
#pragma unroll
            for (int nt = 0; nt < 4; ++nt) {
                const int kb = k0 + nt * 16 + rbase;
#pragma unroll
                for (int r = 0; r < 8; ++r)
                    if (kb + r > qrow) s[nt][r] = -3.0e38f;
            }
        }
        // ---- exp2-based online softmax: mostly lane-local ----
        float x = s[0][0];
#pragma unroll
        for (int nt = 0; nt < 4; ++nt)
#pragma unroll
            for (int r = 0; r < 8; ++r) x = fmaxf(x, s[nt][r]);
        x = fmaxf(x, __shfl_xor(x, 16, 32));          // combine key halves
        const float mnew  = fmaxf(mrun, x);
        const float alpha = __builtin_amdgcn_exp2f(mrun - mnew);
        mrun = mnew;
        float ss = 0.0f;
#pragma unroll
        for (int nt = 0; nt < 4; ++nt)
#pragma unroll
            for (int r = 0; r < 8; ++r) {
                const float p = __builtin_amdgcn_exp2f(s[nt][r] - mnew);
                s[nt][r] = p;
                ss += p;
            }
        ss += __shfl_xor(ss, 16, 32);                 // combine key halves
        lrun = lrun * alpha + ss;
#pragma unroll
        for (int dt = 0; dt < 4; ++dt)
#pragma unroll
            for (int r = 0; r < 8; ++r) accO[dt][r] *= alpha;

        // ---- O^T += V^T * P : P B-frags built in-register from S^T ----
#pragma unroll
        for (int kc = 0; kc < 2; ++kc) {
            // B-frag: lanes 0-15 need keys kc*32+0..15 (subtile t0),
            //         lanes 16-31 need keys kc*32+16..31 (subtile t1).
            // Rows 0-7 of a subtile live in lanes 0-15, rows 8-15 in 16-31.
            v16bf bP;
#pragma unroll
            for (int r = 0; r < 8; ++r) {
                const float s0v  = s[kc * 2 + 0][r];
                const float s1v  = s[kc * 2 + 1][r];
                const float send = lhi ? s0v : s1v;   // value my partner needs
                const float sh   = __shfl_xor(send, 16, 32);
                const float lo   = lhi ? sh : s0v;    // B elems j = r
                const float hi   = lhi ? s1v : sh;    // B elems j = r + 8
                bP[r]     = f2bf(lo);
                bP[r + 8] = f2bf(hi);
            }
            v16bf aV[4];
#pragma unroll
            for (int dt = 0; dt < 4; ++dt) {
                const unsigned short* vp =
                    vt + (size_t)(bh * HD + dt * 16 + lhalf) * Tt
                       + k0 + kc * 32 + lhi * 16;
                aV[dt] = load_bf16_frag(vp, vp + 8);
            }
#pragma unroll
            for (int dt = 0; dt < 4; ++dt)
                accO[dt] = wmma_bf16(aV[dt], bP, accO[dt]);
        }
    }

    // ---- finalize: O^T lane holds 8 consecutive head dims -> packed stores ----
    const float inv = 1.0f / lrun;
    unsigned short* orow = attn + (size_t)(b * Tt + qrow) * Dm + hoff;
#pragma unroll
    for (int dt = 0; dt < 4; ++dt) {
        v8bf pk;
#pragma unroll
        for (int r = 0; r < 8; ++r) pk[r] = f2bf(accO[dt][r] * inv);
        *reinterpret_cast<v8bf*>(orow + dt * 16 + rbase) = pk;
    }
}

// ---------------------------------------------------------------------------
// Launcher
// ---------------------------------------------------------------------------
extern "C" void kernel_launch(void* const* d_in, const int* in_sizes, int n_in,
                              void* d_out, int out_size, void* d_ws, size_t ws_size,
                              hipStream_t stream) {
    (void)in_sizes; (void)n_in; (void)out_size; (void)ws_size;
    const float* q  = (const float*)d_in[0];
    const float* k  = (const float*)d_in[1];
    const float* v  = (const float*)d_in[2];
    const float* Wq = (const float*)d_in[3];
    const float* bq = (const float*)d_in[4];
    const float* Wk = (const float*)d_in[5];
    const float* bk = (const float*)d_in[6];
    const float* Wv = (const float*)d_in[7];
    const float* bv = (const float*)d_in[8];
    const float* Wo = (const float*)d_in[9];
    const float* bo = (const float*)d_in[10];

    char* ws = (char*)d_ws;
    const size_t MB = 1ull << 20;
    unsigned short* Wtq = (unsigned short*)(ws + 0 * MB);    // 2 MB each
    unsigned short* Wtk = (unsigned short*)(ws + 2 * MB);
    unsigned short* Wtv = (unsigned short*)(ws + 4 * MB);
    unsigned short* Wto = (unsigned short*)(ws + 6 * MB);
    unsigned short* qbf = (unsigned short*)(ws + 8 * MB);    // 16 MB each
    unsigned short* kbf = (unsigned short*)(ws + 24 * MB);
    unsigned short* vbf = (unsigned short*)(ws + 40 * MB);
    unsigned short* qwb = (unsigned short*)(ws + 56 * MB);
    unsigned short* kwb = (unsigned short*)(ws + 72 * MB);
    unsigned short* vtb = (unsigned short*)(ws + 88 * MB);
    unsigned short* atb = (unsigned short*)(ws + 104 * MB);  // ends at 120 MB

    dim3 blk(256);

    // activation f32 -> bf16 (once; GEMMs then re-read bf16 8x each)
    const int cgrid = (Mrows * Dm / 8) / 256;
    cvt_kernel<<<cgrid, blk, 0, stream>>>(q, qbf);
    cvt_kernel<<<cgrid, blk, 0, stream>>>(k, kbf);
    cvt_kernel<<<cgrid, blk, 0, stream>>>(v, vbf);

    const int wgrid = (Dm * Dm) / 256;
    wconv_kernel<<<wgrid, blk, 0, stream>>>(Wq, Wtq);
    wconv_kernel<<<wgrid, blk, 0, stream>>>(Wk, Wtk);
    wconv_kernel<<<wgrid, blk, 0, stream>>>(Wv, Wtv);
    wconv_kernel<<<wgrid, blk, 0, stream>>>(Wo, Wto);

    dim3 ggemm(Dm / 128, Mrows / 128);                       // (8, 64)
    // 1/sqrt(HD) * log2(e): exp2-based softmax in the attention kernel
    const float qscale = 0.125f * 1.44269504088896340736f;
    gemm_kernel<0><<<ggemm, blk, 0, stream>>>(qbf, Wtq, bq, qwb, qscale);
    gemm_kernel<0><<<ggemm, blk, 0, stream>>>(kbf, Wtk, bk, kwb, 1.0f);
    gemm_kernel<1><<<ggemm, blk, 0, stream>>>(vbf, Wtv, bv, vtb, 1.0f);

    attn_kernel<<<dim3(Bb * Hh, Tt / 128), blk, 0, stream>>>(qwb, kwb, vtb, atb);

    gemm_kernel<2><<<ggemm, blk, 0, stream>>>(atb, Wto, bo, d_out, 1.0f);
}
